// SwapTestAttention_26345329393865
// MI455X (gfx1250) — matrix operand order repaired and checked by reading, not compiled, not run on hardware
//
#include <hip/hip_runtime.h>
#include <hip/hip_bf16.h>

#define NQ  6
#define NL  2
#define DIM 64   // 2^NQ
#define KD  128  // [re(64) | im(64)]

typedef __attribute__((ext_vector_type(16))) _Float16 v16h;
typedef __attribute__((ext_vector_type(8)))  _Float16 v8h;
typedef __attribute__((ext_vector_type(8)))  float    v8f;

// ---------------------------------------------------------------------------
// Statevector simulation helpers (all loops fully unrolled -> arrays in VGPRs)
// ---------------------------------------------------------------------------
__device__ __forceinline__ void apply_ry(float* re, float* im, float th, int q) {
  float c = cosf(0.5f * th), s = sinf(0.5f * th);
  int bit = 1 << (NQ - 1 - q);
  #pragma unroll
  for (int d = 0; d < DIM; ++d) {
    if (!(d & bit)) {
      int d1 = d | bit;
      float r0 = re[d], r1 = re[d1];
      re[d]  = c * r0 - s * r1;
      re[d1] = s * r0 + c * r1;
      float i0 = im[d], i1 = im[d1];
      im[d]  = c * i0 - s * i1;
      im[d1] = s * i0 + c * i1;
    }
  }
}

__device__ __forceinline__ void apply_rz(float* re, float* im, float th, int q) {
  float c = cosf(0.5f * th), s = sinf(0.5f * th);
  int bit = 1 << (NQ - 1 - q);
  #pragma unroll
  for (int d = 0; d < DIM; ++d) {
    float r = re[d], i = im[d];
    if (d & bit) { re[d] = r * c - i * s; im[d] = i * c + r * s; }  // e^{+i th/2}
    else         { re[d] = r * c + i * s; im[d] = i * c - r * s; }  // e^{-i th/2}
  }
}

__device__ __forceinline__ void simulate(const float* xang, const float* __restrict__ params,
                                         float* re, float* im) {
  #pragma unroll
  for (int d = 0; d < DIM; ++d) { re[d] = 0.f; im[d] = 0.f; }
  re[0] = 1.f;
  #pragma unroll
  for (int q = 0; q < NQ; ++q) apply_ry(re, im, xang[q], q);
  #pragma unroll
  for (int l = 0; l < NL; ++l) {
    #pragma unroll
    for (int q = 0; q < NQ; ++q) {
      apply_ry(re, im, params[(l * NQ + q) * 2 + 0], q);
      apply_rz(re, im, params[(l * NQ + q) * 2 + 1], q);
    }
    #pragma unroll
    for (int q = 0; q < NQ; ++q) {           // CNOT ring q -> (q+1)%NQ
      int cb = 1 << (NQ - 1 - q);
      int tb = 1 << (NQ - 1 - ((q + 1) % NQ));
      #pragma unroll
      for (int d = 0; d < DIM; ++d) {
        if ((d & cb) && !(d & tb)) {
          int d1 = d | tb;
          float tr = re[d]; re[d] = re[d1]; re[d1] = tr;
          float ti = im[d]; im[d] = im[d1]; im[d1] = ti;
        }
      }
    }
  }
}

// ---------------------------------------------------------------------------
// Kernel 1: per-token state prep -> fp16 WMMA operands + V matrix
//   qA[t][0:128] = [ Re(q),  Im(q)]   (dot(qA, kA) = Re<q|k>)
//   qB[t][0:128] = [ Im(q), -Re(q)]   (dot(qB, kA) = Im<q|k>)
//   kA[t][0:128] = [ Re(k),  Im(k)]
//   Vm[t][0:8]   = per-qubit <Z> (6 used, padded to 8 floats)
// Putting the Re/Im split on the A side means the streamed B operand (kA)
// is a single matrix: half the per-iteration load traffic in kernel 2.
// ---------------------------------------------------------------------------
__global__ __launch_bounds__(256) void qprep_kernel(
    const float* __restrict__ x, const float* __restrict__ pq,
    const float* __restrict__ pk, const float* __restrict__ pv,
    _Float16* __restrict__ qA, _Float16* __restrict__ qB,
    _Float16* __restrict__ kA, float* __restrict__ Vm, int n) {
  int t = blockIdx.x * blockDim.x + threadIdx.x;
  if (t >= n) return;
  float xang[NQ];
  #pragma unroll
  for (int q = 0; q < NQ; ++q) xang[q] = x[t * NQ + q];
  float re[DIM], im[DIM];

  simulate(xang, pq, re, im);
  #pragma unroll
  for (int d = 0; d < DIM; ++d) {
    qA[(size_t)t * KD + d]      = (_Float16)re[d];
    qA[(size_t)t * KD + 64 + d] = (_Float16)im[d];
    qB[(size_t)t * KD + d]      = (_Float16)im[d];
    qB[(size_t)t * KD + 64 + d] = (_Float16)(-re[d]);
  }

  simulate(xang, pk, re, im);
  #pragma unroll
  for (int d = 0; d < DIM; ++d) {
    kA[(size_t)t * KD + d]      = (_Float16)re[d];
    kA[(size_t)t * KD + 64 + d] = (_Float16)im[d];
  }

  simulate(xang, pv, re, im);
  float acc[NQ];
  #pragma unroll
  for (int q = 0; q < NQ; ++q) acc[q] = 0.f;
  #pragma unroll
  for (int d = 0; d < DIM; ++d) {
    float p = re[d] * re[d] + im[d] * im[d];
    #pragma unroll
    for (int q = 0; q < NQ; ++q)
      acc[q] += (((d >> (NQ - 1 - q)) & 1) ? -p : p);
  }
  #pragma unroll
  for (int q = 0; q < NQ; ++q) Vm[(size_t)t * 8 + q] = acc[q];
  Vm[(size_t)t * 8 + 6] = 0.f;
  Vm[(size_t)t * 8 + 7] = 0.f;
}

// ---------------------------------------------------------------------------
// Kernel 2: fused flash-style swap-test attention.
// One wave32 handles a 16-query tile; streams all 16-key tiles.
// Per key tile: 8x v_wmma_f32_16x16x32_f16 (4 K-chunks x {Re, Im}); the two
// WMMAs of each chunk share one freshly-loaded B fragment (kA) against the
// two resident A fragments (qA / qB). Then logits = Re^2 + Im^2, online
// softmax, p*V accumulated per lane, reduced across lanes at the end.
// ---------------------------------------------------------------------------
__global__ __launch_bounds__(128) void qattn_kernel(
    const _Float16* __restrict__ qA, const _Float16* __restrict__ qB,
    const _Float16* __restrict__ kA, const float* __restrict__ Vm,
    float* __restrict__ out, int n) {
  const int lane  = threadIdx.x & 31;
  const int wave  = threadIdx.x >> 5;
  const int qtile = blockIdx.x * 4 + wave;
  const int ln    = lane & 15;
  const int hi    = lane >> 4;     // 0: lanes 0-15, 1: lanes 16-31
  const int nkt   = n >> 4;

  // Query A-fragments in contiguous k order (same order as the B loads):
  // lane<16: halves 0-15 <- K kb..kb+15 (kb = kc*32)
  // lane>=16: same +16
  v16h afragR[4], afragI[4];
  {
    const _Float16* qrA = qA + (size_t)(qtile * 16 + ln) * KD;
    const _Float16* qrB = qB + (size_t)(qtile * 16 + ln) * KD;
    #pragma unroll
    for (int kc = 0; kc < 4; ++kc) {
      const int kb = kc * 32 + hi * 16;
      v8h lo = *(const v8h*)(qrA + kb);
      v8h hh = *(const v8h*)(qrA + kb + 8);
      v8h li = *(const v8h*)(qrB + kb);
      v8h hih = *(const v8h*)(qrB + kb + 8);
      #pragma unroll
      for (int e = 0; e < 8; ++e) {
        afragR[kc][e] = lo[e];  afragR[kc][8 + e] = hh[e];
        afragI[kc][e] = li[e];  afragI[kc][8 + e] = hih[e];
      }
    }
  }

  float mrow[8], srow[8], oacc[8][NQ];
  #pragma unroll
  for (int d = 0; d < 8; ++d) {
    mrow[d] = -1e30f; srow[d] = 0.f;
    #pragma unroll
    for (int c = 0; c < NQ; ++c) oacc[d][c] = 0.f;
  }

  for (int kt = 0; kt < nkt; ++kt) {
    const _Float16* ka_row = kA + (size_t)(kt * 16 + ln) * KD;
    if (kt + 1 < nkt)                          // global_prefetch_b8 next key tile
      __builtin_prefetch(ka_row + 16 * KD, 0, 1);

    v8f accR = {}; v8f accI = {};
    #pragma unroll
    for (int kc = 0; kc < 4; ++kc) {
      // B layout: lanes 0-15 hold K=kc*32+0..15, lanes 16-31 K=kc*32+16..31, N=ln
      v16h bA = *(const v16h*)(ka_row + kc * 32 + hi * 16);
      accR = __builtin_amdgcn_wmma_f32_16x16x32_f16(false, afragR[kc], false, bA,
                                                    (short)0, accR, false, false);
      accI = __builtin_amdgcn_wmma_f32_16x16x32_f16(false, afragI[kc], false, bA,
                                                    (short)0, accI, false, false);
    }

    // V of this lane's key
    const float* vrow = Vm + (size_t)(kt * 16 + ln) * 8;
    float vk[NQ];
    #pragma unroll
    for (int c = 0; c < NQ; ++c) vk[c] = vrow[c];

    // C/D layout: VGPR d, lanes 0-15 -> row d; lanes 16-31 -> row d+8; N = ln
    #pragma unroll
    for (int d = 0; d < 8; ++d) {
      float lg = accR[d] * accR[d] + accI[d] * accI[d];  // |<q|k>|^2
      float r = lg;
      #pragma unroll
      for (int o2 = 1; o2 < 16; o2 <<= 1)
        r = fmaxf(r, __shfl_xor(r, o2, 32));             // row max over 16 lanes
      float nm = fmaxf(mrow[d], r);
      float sc = __expf(mrow[d] - nm);
      float p  = __expf(lg - nm);
      mrow[d] = nm;
      srow[d] = srow[d] * sc + p;                        // per-lane partial sum
      #pragma unroll
      for (int c = 0; c < NQ; ++c)
        oacc[d][c] = oacc[d][c] * sc + p * vk[c];        // per-lane partial p*V
    }
  }

  // Final cross-lane reductions + writeback
  #pragma unroll
  for (int d = 0; d < 8; ++d) {
    float s = srow[d];
    float oc[NQ];
    #pragma unroll
    for (int c = 0; c < NQ; ++c) oc[c] = oacc[d][c];
    #pragma unroll
    for (int o2 = 1; o2 < 16; o2 <<= 1) {
      s += __shfl_xor(s, o2, 32);
      #pragma unroll
      for (int c = 0; c < NQ; ++c) oc[c] += __shfl_xor(oc[c], o2, 32);
    }
    if (ln == 0) {
      const int row = qtile * 16 + hi * 8 + d;
      float inv = 1.f / s;
      #pragma unroll
      for (int c = 0; c < NQ; ++c) out[(size_t)row * NQ + c] = oc[c] * inv;
    }
  }
}

// ---------------------------------------------------------------------------
extern "C" void kernel_launch(void* const* d_in, const int* in_sizes, int n_in,
                              void* d_out, int out_size, void* d_ws, size_t ws_size,
                              hipStream_t stream) {
  const float* x  = (const float*)d_in[0];   // (N, 6)
  const float* pq = (const float*)d_in[1];   // (2, 6, 2)
  const float* pk = (const float*)d_in[2];
  const float* pv = (const float*)d_in[3];
  const int n = in_sizes[0] / NQ;            // 8192

  _Float16* qA = (_Float16*)d_ws;
  _Float16* qB = qA + (size_t)n * KD;
  _Float16* kA = qB + (size_t)n * KD;
  float*    Vm = (float*)(kA + (size_t)n * KD);
  float*   outp = (float*)d_out;

  qprep_kernel<<<(n + 255) / 256, 256, 0, stream>>>(x, pq, pk, pv, qA, qB, kA, Vm, n);

  const int qtiles = n / 16;                 // 512 waves, 4 waves per block
  qattn_kernel<<<qtiles / 4, 128, 0, stream>>>(qA, qB, kA, Vm, outp, n);
}
